// W8A8OF16LinearDynamicInputScale_57887569215556
// MI455X (gfx1250) — 8-bit matrix operand order repaired and checked by reading, compiled, not run on hardware
//
#include <hip/hip_runtime.h>
#include <stdint.h>

typedef __attribute__((ext_vector_type(8))) int v8i;
typedef __attribute__((ext_vector_type(4))) int v4i;
typedef __attribute__((ext_vector_type(2))) int v2i;

#define M_DIM 16384
#define N_DIM 4096
#define K_DIM 4096
#define KSTEP 128
#define ROWSTR 144
#define SLABSZ (128 * ROWSTR)
#define NSTEP (K_DIM / KSTEP)

__device__ __forceinline__ void async_b128(uint32_t lds_addr, const int8_t* gaddr) {
    asm volatile("global_load_async_to_lds_b128 %0, %1, off"
                 :: "v"(lds_addr), "v"(gaddr)
                 : "memory");
}

__device__ __forceinline__ void wait_async0() {
#if __has_builtin(__builtin_amdgcn_s_wait_asynccnt)
    __builtin_amdgcn_s_wait_asynccnt(0);
#else
    asm volatile("s_wait_asynccnt 0x0" ::: "memory");
#endif
}

__device__ __forceinline__ void stage_slab(uint32_t lds_slab, const int8_t* gbase,
                                           int k0, int tid) {
    const int row = tid >> 1;
    const int col = (tid & 1) * 64;
    const int8_t* g = gbase + (size_t)row * K_DIM + k0 + col;
    const uint32_t l = lds_slab + (uint32_t)(row * ROWSTR + col);
    async_b128(l,      g);
    async_b128(l + 16, g + 16);
    async_b128(l + 32, g + 32);
    async_b128(l + 48, g + 48);
}

__global__ __launch_bounds__(256) void w8a8_of16_wmma_kernel(
    const int8_t* __restrict__ x,
    const int8_t* __restrict__ w,
    const float*  __restrict__ dq,
    const _Float16* __restrict__ isc,
    const _Float16* __restrict__ bias,
    float* __restrict__ out)
{
    __shared__ __align__(16) int8_t lds[2][2][SLABSZ];

    const int tid  = threadIdx.x;
    const int lane = tid & 31;
    const int wid  = tid >> 5;
    const int half = lane >> 4;
    const int l16  = lane & 15;

    const int wave_m = wid & 3;
    const int wave_n = wid >> 2;

    const int m_blk  = blockIdx.y * 128;
    const int n_blk  = blockIdx.x * 128;
    const int m_base = m_blk + wave_m * 32;
    const int n_base = n_blk + wave_n * 64;

    const int8_t* gA = x + (size_t)m_blk * K_DIM;
    const int8_t* gB = w + (size_t)n_blk * K_DIM;

    const int8_t* curA = &lds[0][0][0];
    const int8_t* curB = &lds[0][1][0];
    const int8_t* nxtA = &lds[1][0][0];
    const int8_t* nxtB = &lds[1][1][0];
    uint32_t curAo = (uint32_t)(uintptr_t)curA;
    uint32_t curBo = (uint32_t)(uintptr_t)curB;
    uint32_t nxtAo = (uint32_t)(uintptr_t)nxtA;
    uint32_t nxtBo = (uint32_t)(uintptr_t)nxtB;

    v8i acc[2][4];
    #pragma unroll
    for (int i = 0; i < 2; ++i)
        #pragma unroll
        for (int j = 0; j < 4; ++j)
            #pragma unroll
            for (int e = 0; e < 8; ++e)
                acc[i][j][e] = 0;

    stage_slab(curAo, gA, 0, tid);
    stage_slab(curBo, gB, 0, tid);

    for (int i = 0; i < NSTEP; ++i) {
        wait_async0();
        __syncthreads();

        if (i + 1 < NSTEP) {
            const int k1 = (i + 1) * KSTEP;
            stage_slab(nxtAo, gA, k1, tid);
            stage_slab(nxtBo, gB, k1, tid);
        }

        #pragma unroll
        for (int kc = 0; kc < 2; ++kc) {
            const int kb = kc * 64;

            v8i a[2];
            #pragma unroll
            for (int mt = 0; mt < 2; ++mt) {
                const int8_t* p = curA + (wave_m * 32 + mt * 16 + l16) * ROWSTR
                                       + kb + 16 * half;
                v2i d0 = *(const v2i*)(p);
                v2i d1 = *(const v2i*)(p + 8);
                v2i d2 = *(const v2i*)(p + 32);
                v2i d3 = *(const v2i*)(p + 40);
                a[mt][0] = d0[0]; a[mt][1] = d0[1];
                a[mt][2] = d1[0]; a[mt][3] = d1[1];
                a[mt][4] = d2[0]; a[mt][5] = d2[1];
                a[mt][6] = d3[0]; a[mt][7] = d3[1];
            }

            v8i b[4];
            #pragma unroll
            for (int nt = 0; nt < 4; ++nt) {
                const int8_t* p = curB + (wave_n * 64 + nt * 16 + l16) * ROWSTR
                                       + kb + 16 * half;
                v4i lo = *(const v4i*)(p);
                v4i hi = *(const v4i*)(p + 32);
                b[nt][0] = lo[0]; b[nt][1] = lo[1]; b[nt][2] = lo[2]; b[nt][3] = lo[3];
                b[nt][4] = hi[0]; b[nt][5] = hi[1]; b[nt][6] = hi[2]; b[nt][7] = hi[3];
            }

            #pragma unroll
            for (int mt = 0; mt < 2; ++mt)
                #pragma unroll
                for (int nt = 0; nt < 4; ++nt)
                    acc[mt][nt] = __builtin_amdgcn_wmma_i32_16x16x64_iu8(
                         true, a[mt],
                         true, b[nt],
                        acc[mt][nt],
                         false,  false);
        }

        __syncthreads();

        const int8_t* tA = curA; curA = nxtA; nxtA = tA;
        const int8_t* tB = curB; curB = nxtB; nxtB = tB;
        uint32_t tAo = curAo; curAo = nxtAo; nxtAo = tAo;
        uint32_t tBo = curBo; curBo = nxtBo; nxtBo = tBo;
    }

    #pragma unroll
    for (int mt = 0; mt < 2; ++mt) {
        const int mrow0 = m_base + mt * 16 + half * 8;
        float is[8];
        #pragma unroll
        for (int g = 0; g < 8; ++g)
            is[g] = (float)isc[mrow0 + g];

        #pragma unroll
        for (int nt = 0; nt < 4; ++nt) {
            const int n = n_base + nt * 16 + l16;
            const float    dqs = (float)(_Float16)dq[n];
            const _Float16 bz  = bias[n];
            #pragma unroll
            for (int g = 0; g < 8; ++g) {
                float v = (float)acc[mt][nt][g] * is[g] * dqs;
                _Float16 h = (_Float16)v + bz;
                out[(size_t)(mrow0 + g) * N_DIM + n] = (float)h;
            }
        }
    }
}

extern "C" void kernel_launch(void* const* d_in, const int* in_sizes, int n_in,
                              void* d_out, int out_size, void* d_ws, size_t ws_size,
                              hipStream_t stream) {
    const int8_t*   x    = (const int8_t*)d_in[0];
    const int8_t*   w    = (const int8_t*)d_in[1];
    const float*    dq   = (const float*)d_in[2];
    const _Float16* isc  = (const _Float16*)d_in[3];
    const _Float16* bias = (const _Float16*)d_in[4];
    float*          out  = (float*)d_out;

    dim3 grid(N_DIM / 128, M_DIM / 128);
    dim3 block(256);
    w8a8_of16_wmma_kernel<<<grid, block, 0, stream>>>(x, w, dq, isc, bias, out);
}
